// GRU_CRF_64321430225394
// MI455X (gfx1250) — matrix operand order repaired and checked by reading, not compiled, not run on hardware
//
#include <hip/hip_runtime.h>
#include <hip/hip_bf16.h>

// ---------------------------------------------------------------------------
// Types / helpers
// ---------------------------------------------------------------------------
typedef __bf16 bf16_t;
typedef __attribute__((ext_vector_type(16))) __bf16 v16bf;
typedef __attribute__((ext_vector_type(8)))  float  v8f;

__device__ __forceinline__ bf16_t f2bf(float f) {
    union { float f; unsigned u; } v; v.f = f;
    unsigned r = v.u + 0x7FFFu + ((v.u >> 16) & 1u);   // round-to-nearest-even
    unsigned short h = (unsigned short)(r >> 16);
    return __builtin_bit_cast(bf16_t, h);
}
__device__ __forceinline__ float bf2f(bf16_t b) {
    unsigned short h = __builtin_bit_cast(unsigned short, b);
    union { unsigned u; float f; } v; v.u = ((unsigned)h) << 16;
    return v.f;
}

__device__ __forceinline__ v8f wmma_bf16(v16bf a, v16bf b, v8f c) {
    // D = A*B + C, f32 accumulate, 16x16x32 bf16
    return __builtin_amdgcn_wmma_f32_16x16x32_bf16(false, a, false, b, (short)0, c, false, false);
}

// CDNA5 async copy: global -> LDS, 16 bytes per lane, tracked by ASYNCcnt.
// GV mode: LDS byte address from the vdst VGPR, global address from a 64-bit VGPR pair (ISA 08 §4.4)
__device__ __forceinline__ void async_load_b128(unsigned lds_byte_off, const void* gptr) {
    asm volatile("global_load_async_to_lds_b128 %0, %1, off"
                 :: "v"(lds_byte_off), "v"((unsigned long long)gptr)
                 : "memory");
}
__device__ __forceinline__ void wait_asynccnt0() {
    asm volatile("s_wait_asynccnt 0x0" ::: "memory");
}

// A-fragment (16x32, bf16) from row-major bf16 matrix (global or LDS).
// ISA 16-bit A layout: lane L: M = L%16; elems 0..7 -> K = base..base+7,
// elems 8..15 -> K = base+16..base+23, base = (L<16 ? 0 : 8).
__device__ __forceinline__ v16bf loadA_rm(const bf16_t* A, int lda, size_t m0, int k0, int lane) {
    size_t m  = m0 + (size_t)(lane & 15);
    int    kb = k0 + ((lane & 16) ? 8 : 0);
    const bf16_t* p = A + m * (size_t)lda + kb;
    union { uint4 u[2]; v16bf v; } f;
    f.u[0] = *(const uint4*)(p);
    f.u[1] = *(const uint4*)(p + 16);
    return f.v;
}

// B-fragment from pre-swizzled fragment buffer: tile = 512 bf16, lane-contiguous.
__device__ __forceinline__ v16bf loadB_frag(const bf16_t* fragBase, int tileIdx, int lane) {
    const uint4* p = (const uint4*)(fragBase + (size_t)tileIdx * 512 + lane * 16);
    union { uint4 u[2]; v16bf v; } f;
    f.u[0] = p[0];
    f.u[1] = p[1];
    return f.v;
}

// ---------------------------------------------------------------------------
// Kernel: f32 -> bf16 activation copy with K padding (zeros beyond E)
// ---------------------------------------------------------------------------
__global__ void conv_pad_bf16(const float* __restrict__ x, bf16_t* __restrict__ dst,
                              int E, int ldp, size_t rows) {
    size_t i = (size_t)blockIdx.x * blockDim.x + threadIdx.x;
    size_t total = rows * (size_t)ldp;
    if (i >= total) return;
    size_t row = i / ldp;
    int    col = (int)(i % ldp);
    dst[i] = f2bf(col < E ? x[row * (size_t)E + col] : 0.f);
}

// ---------------------------------------------------------------------------
// Kernel: pre-swizzle weight W[N,K] (row-major f32) into WMMA B-fragments.
// B layout (ISA 7.12.2): lane L: N = L%16; elem j -> K = 8*(L>=16) + (j<8 ? j : j+8).
// Tile order: [ntile][ktile][lane][16].  Zero-padded for n>=N or k>=K.
// ---------------------------------------------------------------------------
__global__ void prep_wfrag(const float* __restrict__ W, bf16_t* __restrict__ dst,
                           int N, int K, int Npad, int Kpad) {
    int tid = blockIdx.x * blockDim.x + threadIdx.x;
    int nKt = Kpad >> 5;
    int total = (Npad >> 4) * nKt * 32;
    if (tid >= total) return;
    int lane = tid & 31;
    int tile = tid >> 5;
    int kt = tile % nKt;
    int nt = tile / nKt;
    int n     = nt * 16 + (lane & 15);
    int kbase = kt * 32 + ((lane & 16) ? 8 : 0);
    bf16_t* out = dst + (size_t)tile * 512 + lane * 16;
#pragma unroll
    for (int j = 0; j < 16; ++j) {
        int k = kbase + j + ((j >= 8) ? 8 : 0);
        float v = (n < N && k < K) ? W[(size_t)n * K + k] : 0.f;
        out[j] = f2bf(v);
    }
}

// ---------------------------------------------------------------------------
// Kernel: C[M,N] = A[M,Kpad](bf16) @ Wfrag^T + bias ;  one wave -> 32x32 tile
// ---------------------------------------------------------------------------
__global__ void gemm_bias_kernel(const bf16_t* __restrict__ A, int lda,
                                 const bf16_t* __restrict__ Bfrag, int nKt,
                                 const float* __restrict__ bias, int biasN,
                                 float* __restrict__ C, int ldc,
                                 size_t M, int N) {
    int lane = threadIdx.x & 31;
    size_t wave = ((size_t)blockIdx.x * blockDim.x + threadIdx.x) >> 5;
    int nW = N / 32;
    size_t m0 = (wave / (size_t)nW) * 32;
    int    n0 = (int)(wave % (size_t)nW) * 32;
    if (m0 >= M) return;

    v8f c00 = {0.f,0.f,0.f,0.f,0.f,0.f,0.f,0.f};
    v8f c01 = c00, c10 = c00, c11 = c00;
    int nt0 = n0 >> 4, nt1 = nt0 + 1;

    for (int kt = 0; kt < nKt; ++kt) {
        if (kt + 1 < nKt) {  // prefetch next A strip (global_prefetch_b8)
            __builtin_prefetch(A + (m0 + (size_t)(lane & 15)) * (size_t)lda + (kt + 1) * 32, 0, 1);
        }
        v16bf a0 = loadA_rm(A, lda, m0,      kt * 32, lane);
        v16bf a1 = loadA_rm(A, lda, m0 + 16, kt * 32, lane);
        v16bf b0 = loadB_frag(Bfrag, nt0 * nKt + kt, lane);
        v16bf b1 = loadB_frag(Bfrag, nt1 * nKt + kt, lane);
        c00 = wmma_bf16(a0, b0, c00);
        c01 = wmma_bf16(a0, b1, c01);
        c10 = wmma_bf16(a1, b0, c10);
        c11 = wmma_bf16(a1, b1, c11);
    }

    // D layout: row = r + 8*(lane>=16), col = lane%16
    int colL = lane & 15;
    int rowH = (lane & 16) ? 8 : 0;
#pragma unroll
    for (int q = 0; q < 4; ++q) {
        v8f c = (q == 0) ? c00 : (q == 1) ? c01 : (q == 2) ? c10 : c11;
        size_t mb = m0 + ((q >= 2) ? 16 : 0) + rowH;
        int    cb = n0 + ((q & 1) ? 16 : 0) + colL;
        float  bv = (cb < biasN) ? bias[cb] : 0.f;
#pragma unroll
        for (int r = 0; r < 8; ++r)
            C[(mb + r) * (size_t)ldc + cb] = c[r] + bv;
    }
}

// ---------------------------------------------------------------------------
// Kernel: GRU scan, one persistent workgroup per direction (grid = 2).
//   gh = h @ Whh^T + bhh   (WMMA, Whh fragments resident in LDS)
//   G[t] slab (64x384 f32) is async-copied global->LDS one step ahead so the
//   HBM latency overlaps the WMMA GEMM on the serial chain (ASYNCcnt).
// LDS: whh 96K | hbuf 16K | ghb 96K | ghG 96K  = 304 KB (<320 KB/WG)
// ---------------------------------------------------------------------------
#define GRU_OFF_WHH  0
#define GRU_OFF_HBUF 98304
#define GRU_OFF_GHB  114688
#define GRU_OFF_GHG  212992
#define GRU_LDS_BYTES (212992 + 98304)

__global__ void gru_scan_kernel(const float* __restrict__ Gf, const float* __restrict__ Gb,
                                const bf16_t* __restrict__ WhhF, const bf16_t* __restrict__ WhhB,
                                const float* __restrict__ bhhF, const float* __restrict__ bhhB,
                                bf16_t* __restrict__ Hout, int S) {
    extern __shared__ char smem[];
    bf16_t* whh  = (bf16_t*)(smem + GRU_OFF_WHH);    // 24*4 tiles * 512 bf16 = 96 KB
    bf16_t* hbuf = (bf16_t*)(smem + GRU_OFF_HBUF);   // 64*128 bf16 = 16 KB
    float*  ghb  = (float*) (smem + GRU_OFF_GHB);    // 64*384 f32  = 96 KB
    float*  ghG  = (float*) (smem + GRU_OFF_GHG);    // staged G[t] slab, 96 KB

    int dir = blockIdx.x;
    const float*  G    = dir ? Gb   : Gf;
    const bf16_t* WF   = dir ? WhhB : WhhF;
    const float*  bhh  = dir ? bhhB : bhhF;
    int dirOff = dir ? 128 : 0;
    int tid = threadIdx.x;            // 256 threads = 8 waves
    int lane = tid & 31, wave = tid >> 5;

    // Issue async global->LDS copies for token tt's G slab (6144 x b128).
    auto issueG = [&](int tt) {
        for (int i = tid; i < 6144; i += 256) {
            int bidx = i / 96;                 // batch row
            int col4 = (i % 96) * 4;           // float column (16B chunk)
            const float* gp = G + ((size_t)bidx * S + tt) * 384 + col4;
            async_load_b128((unsigned)(GRU_OFF_GHG + i * 16), gp);
        }
    };

    { // stage Whh fragments into LDS, zero h state
        const uint4* src = (const uint4*)WF;
        uint4* dst = (uint4*)whh;
        for (int i = tid; i < 6144; i += 256) dst[i] = src[i];
        for (int i = tid; i < 8192; i += 256) hbuf[i] = f2bf(0.f);
    }
    // prologue: start fetching G for the first token
    issueG(dir ? (S - 1) : 0);
    __syncthreads();

    for (int t = 0; t < S; ++t) {
        int tt = dir ? (S - 1 - t) : t;

        // --- gh = h @ Whh^T : M=64, N=384, K=128.  Wave w owns N-tiles [3w,3w+3).
        for (int mt = 0; mt < 4; ++mt) {
            v8f c0 = {0.f,0.f,0.f,0.f,0.f,0.f,0.f,0.f};
            v8f c1 = c0, c2 = c0;
#pragma unroll
            for (int kt = 0; kt < 4; ++kt) {
                v16bf a  = loadA_rm(hbuf, 128, (size_t)(mt * 16), kt * 32, lane);
                v16bf b0 = loadB_frag(whh, (wave * 3 + 0) * 4 + kt, lane);
                v16bf b1 = loadB_frag(whh, (wave * 3 + 1) * 4 + kt, lane);
                v16bf b2 = loadB_frag(whh, (wave * 3 + 2) * 4 + kt, lane);
                c0 = wmma_bf16(a, b0, c0);
                c1 = wmma_bf16(a, b1, c1);
                c2 = wmma_bf16(a, b2, c2);
            }
            int row0 = mt * 16 + ((lane & 16) ? 8 : 0);
            int col  = lane & 15;
            int nb   = wave * 48;
#pragma unroll
            for (int r = 0; r < 8; ++r) {
                ghb[(row0 + r) * 384 + nb +      col] = c0[r];
                ghb[(row0 + r) * 384 + nb + 16 + col] = c1[r];
                ghb[(row0 + r) * 384 + nb + 32 + col] = c2[r];
            }
        }
        wait_asynccnt0();       // this wave's G[t] chunks landed in LDS
        __syncthreads();        // => all waves' chunks + ghb visible

        // --- gates: h_new = (1-z)*n + z*h   (G read from staged LDS slab)
        for (int i = tid; i < 8192; i += 256) {
            int bidx = i >> 7, hh = i & 127;
            float ir  = ghG[bidx * 384 +       hh];
            float iz  = ghG[bidx * 384 + 128 + hh];
            float in_ = ghG[bidx * 384 + 256 + hh];
            float hr = ghb[bidx * 384 +       hh] + bhh[hh];
            float hz = ghb[bidx * 384 + 128 + hh] + bhh[128 + hh];
            float hn = ghb[bidx * 384 + 256 + hh] + bhh[256 + hh];
            float r = 1.f / (1.f + __expf(-(ir + hr)));
            float z = 1.f / (1.f + __expf(-(iz + hz)));
            float n = tanhf(in_ + r * hn);
            float hprev = bf2f(hbuf[i]);
            float hnew = (1.f - z) * n + z * hprev;
            hbuf[i] = f2bf(hnew);
            Hout[((size_t)bidx * S + tt) * 256 + dirOff + hh] = f2bf(hnew);
        }
        __syncthreads();        // ghG fully consumed, hbuf stable

        if (t + 1 < S) issueG(dir ? (S - 2 - t) : (t + 1));   // overlap next fetch
    }
}

// ---------------------------------------------------------------------------
// CRF: one block (1 wave32) per batch row. Thread j keeps alpha[j] (j<30).
// ---------------------------------------------------------------------------
__global__ void zero_out_kernel(float* o) { if (threadIdx.x == 0 && blockIdx.x == 0) o[0] = 0.f; }

__global__ void crf_kernel(const float* __restrict__ emis /*[B*S,32]*/,
                           const unsigned char* __restrict__ mask,
                           const int* __restrict__ tags,
                           const float* __restrict__ trans,
                           const float* __restrict__ start_t,
                           const float* __restrict__ end_t,
                           float* __restrict__ out, int S) {
    const float NEG = -3.402823466e38f;
    int b = blockIdx.x;
    int j = threadIdx.x;                 // 32 threads
    __shared__ float alpha[32];
    __shared__ float tr[900];
    __shared__ float red[32];
    for (int i = j; i < 900; i += 32) tr[i] = trans[i];
    const float* eb = emis + (size_t)b * S * 32;
    if (j < 30) alpha[j] = start_t[j] + eb[j];
    __syncthreads();

    for (int t = 1; t < S; ++t) {
        float na = 0.f;
        bool m = mask[(size_t)b * S + t] != 0;
        if (j < 30) {
            float mx = NEG;
            for (int i = 0; i < 30; ++i) mx = fmaxf(mx, alpha[i] + tr[i * 30 + j]);
            float s = 0.f;
            for (int i = 0; i < 30; ++i) s += __expf(alpha[i] + tr[i * 30 + j] - mx);
            na = mx + __logf(s) + eb[(size_t)t * 32 + j];
        }
        __syncthreads();
        if (j < 30 && m) alpha[j] = na;
        __syncthreads();
    }

    if (j < 30) red[j] = alpha[j] + end_t[j]; else red[j] = NEG;
    __syncthreads();
    if (j == 0) {
        float mx = NEG;
        for (int i = 0; i < 30; ++i) mx = fmaxf(mx, red[i]);
        float s = 0.f;
        for (int i = 0; i < 30; ++i) s += __expf(red[i] - mx);
        float logZ = mx + __logf(s);

        const int* tg = tags + (size_t)b * S;
        const unsigned char* mk = mask + (size_t)b * S;
        float score = start_t[tg[0]] + eb[tg[0]];
        int cnt = mk[0] ? 1 : 0;
        for (int t = 1; t < S; ++t) {
            if (mk[t]) {
                score += tr[tg[t - 1] * 30 + tg[t]] + eb[(size_t)t * 32 + tg[t]];
                ++cnt;
            }
        }
        int last = cnt - 1;
        score += end_t[tg[last]];
        atomicAdd(out, logZ - score);
    }
}

// ---------------------------------------------------------------------------
// Host launcher
// ---------------------------------------------------------------------------
extern "C" void kernel_launch(void* const* d_in, const int* in_sizes, int n_in,
                              void* d_out, int out_size, void* d_ws, size_t ws_size,
                              hipStream_t stream) {
    (void)in_sizes; (void)n_in; (void)out_size; (void)ws_size;

    const int B = 64, S = 1024, E = 300;
    const size_t Mtok = (size_t)B * S;           // 65536 tokens

    const float* x      = (const float*)d_in[0];
    const unsigned char* mask = (const unsigned char*)d_in[1];
    const int*   target = (const int*)d_in[2];
    const float* w_ih_l0f = (const float*)d_in[3];
    const float* w_hh_l0f = (const float*)d_in[4];
    const float* b_ih_l0f = (const float*)d_in[5];
    const float* b_hh_l0f = (const float*)d_in[6];
    const float* w_ih_l0b = (const float*)d_in[7];
    const float* w_hh_l0b = (const float*)d_in[8];
    const float* b_ih_l0b = (const float*)d_in[9];
    const float* b_hh_l0b = (const float*)d_in[10];
    const float* w_ih_l1f = (const float*)d_in[11];
    const float* w_hh_l1f = (const float*)d_in[12];
    const float* b_ih_l1f = (const float*)d_in[13];
    const float* b_hh_l1f = (const float*)d_in[14];
    const float* w_ih_l1b = (const float*)d_in[15];
    const float* w_hh_l1b = (const float*)d_in[16];
    const float* b_ih_l1b = (const float*)d_in[17];
    const float* b_hh_l1b = (const float*)d_in[18];
    const float* w_tag   = (const float*)d_in[19];
    const float* b_tag   = (const float*)d_in[20];
    const float* trans   = (const float*)d_in[21];
    const float* start_t = (const float*)d_in[22];
    const float* end_t   = (const float*)d_in[23];

    // ---- workspace carve ----
    char* p = (char*)d_ws;
    auto alloc = [&](size_t bytes) -> char* {
        char* r = p;
        p += (bytes + 255) & ~(size_t)255;
        return r;
    };
    bf16_t* A0   = (bf16_t*)alloc(Mtok * 320 * 2);   // layer0 input, K padded 300->320
    bf16_t* A1   = (bf16_t*)alloc(Mtok * 256 * 2);   // layer0 output / layer1 input
    bf16_t* A2   = (bf16_t*)alloc(Mtok * 256 * 2);   // layer1 output
    float*  Gf   = (float*) alloc(Mtok * 384 * 4);   // per-dir gate pre-activations (reused)
    float*  Gb   = (float*) alloc(Mtok * 384 * 4);
    float*  Emis = (float*) alloc(Mtok * 32 * 4);
    bf16_t* Fih0f = (bf16_t*)alloc((size_t)24 * 10 * 512 * 2);
    bf16_t* Fih0b = (bf16_t*)alloc((size_t)24 * 10 * 512 * 2);
    bf16_t* Fhh0f = (bf16_t*)alloc((size_t)24 * 4 * 512 * 2);
    bf16_t* Fhh0b = (bf16_t*)alloc((size_t)24 * 4 * 512 * 2);
    bf16_t* Fih1f = (bf16_t*)alloc((size_t)24 * 8 * 512 * 2);
    bf16_t* Fih1b = (bf16_t*)alloc((size_t)24 * 8 * 512 * 2);
    bf16_t* Fhh1f = (bf16_t*)alloc((size_t)24 * 4 * 512 * 2);
    bf16_t* Fhh1b = (bf16_t*)alloc((size_t)24 * 4 * 512 * 2);
    bf16_t* Ftag  = (bf16_t*)alloc((size_t)2 * 8 * 512 * 2);

    // ---- 1) x -> bf16 padded ----
    {
        size_t total = Mtok * 320;
        int blocks = (int)((total + 255) / 256);
        conv_pad_bf16<<<blocks, 256, 0, stream>>>(x, A0, E, 320, Mtok);
    }

    // ---- 2) weight fragment pre-swizzle ----
    auto prep = [&](const float* W, bf16_t* dst, int N, int K, int Npad, int Kpad) {
        int total = (Npad / 16) * (Kpad / 32) * 32;
        prep_wfrag<<<(total + 255) / 256, 256, 0, stream>>>(W, dst, N, K, Npad, Kpad);
    };
    prep(w_ih_l0f, Fih0f, 384, 300, 384, 320);
    prep(w_ih_l0b, Fih0b, 384, 300, 384, 320);
    prep(w_hh_l0f, Fhh0f, 384, 128, 384, 128);
    prep(w_hh_l0b, Fhh0b, 384, 128, 384, 128);
    prep(w_ih_l1f, Fih1f, 384, 256, 384, 256);
    prep(w_ih_l1b, Fih1b, 384, 256, 384, 256);
    prep(w_hh_l1f, Fhh1f, 384, 128, 384, 128);
    prep(w_hh_l1b, Fhh1b, 384, 128, 384, 128);
    prep(w_tag,   Ftag,    30, 256,  32, 256);

    // ---- GEMM launch helper: M x N, one wave per 32x32 tile ----
    auto gemm = [&](const bf16_t* A, int lda, const bf16_t* Bfrag, int nKt,
                    const float* bias, int biasN, float* C, int ldc, int N) {
        size_t waves = (Mtok / 32) * (size_t)(N / 32);
        size_t threads = waves * 32;
        int blocks = (int)((threads + 255) / 256);
        gemm_bias_kernel<<<blocks, 256, 0, stream>>>(A, lda, Bfrag, nKt, bias, biasN,
                                                     C, ldc, Mtok, N);
    };

    // ---- 3) layer-0 input projections ----
    gemm(A0, 320, Fih0f, 10, b_ih_l0f, 384, Gf, 384, 384);
    gemm(A0, 320, Fih0b, 10, b_ih_l0b, 384, Gb, 384, 384);

    // ---- 4) layer-0 recurrent scan (1 WG per direction) ----
    gru_scan_kernel<<<2, 256, GRU_LDS_BYTES, stream>>>(Gf, Gb, Fhh0f, Fhh0b,
                                                       b_hh_l0f, b_hh_l0b, A1, S);

    // ---- 5) layer-1 input projections ----
    gemm(A1, 256, Fih1f, 8, b_ih_l1f, 384, Gf, 384, 384);
    gemm(A1, 256, Fih1b, 8, b_ih_l1b, 384, Gb, 384, 384);

    // ---- 6) layer-1 recurrent scan ----
    gru_scan_kernel<<<2, 256, GRU_LDS_BYTES, stream>>>(Gf, Gb, Fhh1f, Fhh1b,
                                                       b_hh_l1f, b_hh_l1b, A2, S);

    // ---- 7) emissions: [M,256] x [256,32] + b_tag (cols >=30 are zero pad) ----
    gemm(A2, 256, Ftag, 8, b_tag, 30, Emis, 32, 32);

    // ---- 8) CRF NLL -> scalar ----
    zero_out_kernel<<<1, 1, 0, stream>>>((float*)d_out);
    crf_kernel<<<B, 32, 0, stream>>>(Emis, mask, target, trans, start_t, end_t,
                                     (float*)d_out, S);
}
